// LLAttentionBlock_72310069395744
// MI455X (gfx1250) — matrix operand order repaired and checked by reading, not compiled, not run on hardware
//
#include <hip/hip_runtime.h>
#include <hip/hip_bf16.h>

// ---------------------------------------------------------------------------
// LLAttentionBlock for MI455X (gfx1250, wave32, WMMA 16x16x32 f16 + TDM)
// B=2 N=2048 D=1024 H=16 HD=64 CHUNK=128
// ---------------------------------------------------------------------------

#define Bb   2
#define Nn   2048
#define Dd   1024
#define Hh   16
#define HD_  64
#define BN   (Bb * Nn)          // 4096 rows
#define WCAT 5120               // q(1024) k(1024) v(1024) kk(1024) vk(1024)

typedef _Float16 v16h __attribute__((ext_vector_type(16)));
typedef _Float16 v8h  __attribute__((ext_vector_type(8)));
typedef float    v8f  __attribute__((ext_vector_type(8)));

__device__ __forceinline__ v8f wmma16(v16h a, v16h b, v8f c) {
    // D = A(16x32 f16) x B(32x16 f16) + C(16x16 f32)
    return __builtin_amdgcn_wmma_f32_16x16x32_f16(
        /*neg_a=*/false, a, /*neg_b=*/false, b,
        /*c_mod=*/(short)0, c, /*reuse_a=*/false, /*reuse_b=*/false);
}

// Build a v16h A/B fragment from two 8-half contiguous chunks (16B aligned).
__device__ __forceinline__ v16h ldfrag_split(const _Float16* p0, const _Float16* p1) {
    v8h a = *(const v8h*)p0;
    v8h b = *(const v8h*)p1;
    v16h r;
#pragma unroll
    for (int i = 0; i < 8; ++i) { r[i] = a[i]; r[8 + i] = b[i]; }
    return r;
}
__device__ __forceinline__ v16h ldfrag_cont(const _Float16* p) {
    return ldfrag_split(p, p + 8);
}

// ---------------------------------------------------------------------------
// Tensor Data Mover: async 2D tile load Global -> LDS (TENSORcnt-tracked).
// D# packed per CDNA5 ISA ch.8: group0 = {flags, lds_addr, global_addr, type=2},
// group1 = {data_size/pad, tensor dims, tile dims, dim0 stride}.
// This toolchain (clang-23 / therock-10.0) exposes the 6-arg builtin:
//   (u32x4 g0, i32x8 g1, i32x4 g2, i32x4 g3, i32x8 extra, i32 cpol)
// ---------------------------------------------------------------------------
#if defined(__has_builtin)
#if __has_builtin(__builtin_amdgcn_tensor_load_to_lds) && \
    __has_builtin(__builtin_amdgcn_s_wait_tensorcnt)
#define HAVE_TDM 1
#endif
#endif
#ifndef HAVE_TDM
#define HAVE_TDM 0
#endif

#if HAVE_TDM
typedef unsigned int u32x4 __attribute__((ext_vector_type(4)));
typedef int          i32x8 __attribute__((ext_vector_type(8)));
typedef int          i32x4 __attribute__((ext_vector_type(4)));

// Load a tile_d1 x tile_d0 (rows x elements, 2-byte elems) tile whose rows are
// stride_d0 elements apart in global memory into LDS at lds_addr, inserting
// pad_amount dwords of padding after every pad_interval dwords (row padding).
__device__ __forceinline__ void tdm_load_2d_f16(
    unsigned lds_addr, const void* gaddr,
    unsigned tensor_d0, unsigned tensor_d1, unsigned long long stride_d0,
    unsigned tile_d0, unsigned tile_d1,
    unsigned pad_interval_code, unsigned pad_amount_code) {
    unsigned long long ga = (unsigned long long)(uintptr_t)gaddr;
    u32x4 g0;
    g0[0] = 1u;                                        // count=1, user mode
    g0[1] = lds_addr;                                  // LDS byte address
    g0[2] = (unsigned)(ga & 0xffffffffu);              // global_addr[31:0]
    g0[3] = (unsigned)((ga >> 32) & 0x01ffffffu)       // global_addr[56:32]
          | (2u << 30);                                // type = 2 ("image")
    i32x8 g1;
    g1[0] = (int)((1u << 16)                           // data_size = 2 bytes
                | (1u << 20)                           // pad_enable
                | (pad_interval_code << 22)
                | (pad_amount_code << 25));
    g1[1] = (int)((tensor_d0 & 0xffffu) << 16);        // tensor_dim0[15:0]
    g1[2] = (int)(((tensor_d0 >> 16) & 0xffffu)        // tensor_dim0[31:16]
                | ((tensor_d1 & 0xffffu) << 16));      // tensor_dim1[15:0]
    g1[3] = (int)(((tensor_d1 >> 16) & 0xffffu)        // tensor_dim1[31:16]
                | ((tile_d0 & 0xffffu) << 16));        // tile_dim0
    g1[4] = (int)(tile_d1 & 0xffffu);                  // tile_dim1 (tile_dim2=0)
    g1[5] = (int)(stride_d0 & 0xffffffffull);          // tensor_dim0_stride lo
    g1[6] = (int)((stride_d0 >> 32) & 0xffffull);      // stride hi (dim1_stride=0)
    g1[7] = 0;
    i32x4 z4 = {0, 0, 0, 0};
    i32x8 z8 = {0, 0, 0, 0, 0, 0, 0, 0};
    __builtin_amdgcn_tensor_load_to_lds(g0, g1, z4, z4, z8, 0);
}
#endif

// ---------------------------------------------------------------------------
// Kernel 1: fp32 -> fp16 conversion / weight packing
//   Xh  [4096 x 1024]
//   Wh  [1024 x 5120] = [w_qkv | w_kvkv[:, :2048]]
//   Wo  [1024 x 1024]
// ---------------------------------------------------------------------------
#define N_XH  (BN * Dd)                 // 4,194,304
#define N_WH  (Dd * WCAT)               // 5,242,880
#define N_WO  (Dd * Dd)                 // 1,048,576
#define N_TOT (N_XH + N_WH + N_WO)      // 10,485,760

__global__ void convert_pack_kernel(const float* __restrict__ X,
                                    const float* __restrict__ wqkv,
                                    const float* __restrict__ wkvkv,
                                    const float* __restrict__ wout,
                                    _Float16* __restrict__ Xh,
                                    _Float16* __restrict__ Wh,
                                    _Float16* __restrict__ Wo) {
    for (int idx = blockIdx.x * blockDim.x + threadIdx.x; idx < N_TOT;
         idx += gridDim.x * blockDim.x) {
        if (idx < N_XH) {
            Xh[idx] = (_Float16)X[idx];
        } else if (idx < N_XH + N_WH) {
            int j = idx - N_XH;
            int k = j / WCAT, c = j - k * WCAT;
            float v = (c < 3072) ? wqkv[k * 3072 + c] : wkvkv[k * 4096 + (c - 3072)];
            Wh[j] = (_Float16)v;
        } else {
            int j = idx - (N_XH + N_WH);
            Wo[j] = (_Float16)wout[j];
        }
    }
}

// ---------------------------------------------------------------------------
// Kernel 2: generic WMMA GEMM  C[MxN] = A[MxK](f16) * B[KxN](f16) (+bias)
// Block tile 128x128, BK=32, 256 threads = 8 waves (4x2), wave tile 32x64.
// A tile staged by the TDM (async tensor load, row padding via D#);
// B tile staged manually (transposed). Requires M%128, N%128, K%32 == 0.
// ---------------------------------------------------------------------------
template <bool OUT_HALF, bool BIAS>
__global__ __launch_bounds__(256) void gemm_wmma_kernel(
    const _Float16* __restrict__ A, const _Float16* __restrict__ B,
    void* __restrict__ Cout, const float* __restrict__ bias,
    int M, int N, int K, int lda, int ldb, int ldc) {
    __shared__ _Float16 As[128][40];   // [m][k], 32 data + 8 pad halves per row
    __shared__ _Float16 Bst[128][40];  // [n][k] (transposed stage)

    const int t  = threadIdx.x;
    const int l  = t & 31;
    const int w  = t >> 5;
    const int wm = w >> 1;     // 0..3
    const int wn = w & 1;      // 0..1
    const int lh = l >> 4;     // half-wave id
    const int ll = l & 15;

    const int rowBase = blockIdx.y * 128;
    const int colBase = blockIdx.x * 128;

    v8f acc[2][4];
#pragma unroll
    for (int mt = 0; mt < 2; ++mt)
#pragma unroll
        for (int nt = 0; nt < 4; ++nt)
#pragma unroll
            for (int r = 0; r < 8; ++r) acc[mt][nt][r] = 0.0f;

    const int bk = t >> 3, bc0 = (t & 7) * 16;     // B stage coords
#if !HAVE_TDM
    const int arow = t >> 1, ak0 = (t & 1) * 16;   // A stage coords (fallback)
#endif

    for (int kb = 0; kb < K; kb += 32) {
        // --- stage A tile: async TDM 128x32-half tile with 8-half row pad ---
#if HAVE_TDM
        if (w == 0) {
            tdm_load_2d_f16((unsigned)(uintptr_t)&As[0][0],
                            A + (size_t)rowBase * lda + kb,
                            /*tensor_d0=*/(unsigned)K, /*tensor_d1=*/(unsigned)M,
                            /*stride_d0=*/(unsigned long long)lda,
                            /*tile_d0=*/32, /*tile_d1=*/128,
                            /*pad_interval: 16 dwords ->*/3,
                            /*pad_amount:   4 dwords ->*/3);
        }
#else
        {
            const _Float16* src = A + (size_t)(rowBase + arow) * lda + kb + ak0;
            *(v8h*)&As[arow][ak0]     = *(const v8h*)src;
            *(v8h*)&As[arow][ak0 + 8] = *(const v8h*)(src + 8);
        }
#endif
        // --- stage B tile transposed ---
        {
            const _Float16* src = B + (size_t)(kb + bk) * ldb + colBase + bc0;
            v8h b0 = *(const v8h*)src;
            v8h b1 = *(const v8h*)(src + 8);
#pragma unroll
            for (int j = 0; j < 8; ++j) {
                Bst[bc0 + j][bk]     = b0[j];
                Bst[bc0 + 8 + j][bk] = b1[j];
            }
            if (kb + 32 < K)  // prefetch next K tile (global_prefetch_b8)
                __builtin_prefetch(src + 32 * ldb, 0, 1);
        }
#if HAVE_TDM
        if (w == 0) __builtin_amdgcn_s_wait_tensorcnt(0);
#endif
        __syncthreads();

        // --- fragments + 8 WMMAs ---
        v16h af[2], bf[4];
#pragma unroll
        for (int mt = 0; mt < 2; ++mt) {
            const _Float16* p = &As[wm * 32 + mt * 16 + ll][lh * 8];
            af[mt] = ldfrag_split(p, p + 16);   // K chunks {0..7,16..23}/{8..15,24..31}
        }
#pragma unroll
        for (int nt = 0; nt < 4; ++nt) {
            const _Float16* p = &Bst[wn * 64 + nt * 16 + ll][lh * 8];
            bf[nt] = ldfrag_split(p, p + 16);   // K chunks {0..7,16..23}/{8..15,24..31}
        }
#pragma unroll
        for (int mt = 0; mt < 2; ++mt)
#pragma unroll
            for (int nt = 0; nt < 4; ++nt)
                acc[mt][nt] = wmma16(af[mt], bf[nt], acc[mt][nt]);
        __syncthreads();
    }

    // --- epilogue: C layout row = r + 8*lh, col = ll ---
#pragma unroll
    for (int mt = 0; mt < 2; ++mt)
#pragma unroll
        for (int nt = 0; nt < 4; ++nt)
#pragma unroll
            for (int r = 0; r < 8; ++r) {
                int row = rowBase + wm * 32 + mt * 16 + r + 8 * lh;
                int col = colBase + wn * 64 + nt * 16 + ll;
                float v = acc[mt][nt][r];
                if (BIAS) v += bias[col];
                if (OUT_HALF)
                    ((_Float16*)Cout)[(size_t)row * ldc + col] = (_Float16)v;
                else
                    ((float*)Cout)[(size_t)row * ldc + col] = v;
            }
}

// ---------------------------------------------------------------------------
// Kernel 3: token-serial linear-attention scan (chunked cumsum == running sum)
// One wave (32 threads) per (b,h); each lane owns 2 rows of the 64x64 state
// entirely in VGPRs. Frobenius norm via wave shfl_xor. Output Qn f16,
// head-major [B,H,N,64].
// ---------------------------------------------------------------------------
__global__ __launch_bounds__(32) void chunked_scan_kernel(
    const _Float16* __restrict__ P, _Float16* __restrict__ Qn) {
    __shared__ float kks[64], vks[64], qs[64];
    const int bh = blockIdx.x;       // b*16 + h
    const int l  = threadIdx.x;      // 0..31
    const int b  = bh >> 4, h = bh & 15;

    float S0[64], S1[64];            // rows j=l and j=l+32 of the state
#pragma unroll
    for (int i = 0; i < 64; ++i) {
        S0[i] = (i == l)      ? 8.0f : 0.0f;   // sqrt(64)*I
        S1[i] = (i == l + 32) ? 8.0f : 0.0f;
    }
    const int qcol = h * 64;
    for (int n = 0; n < Nn; ++n) {
        size_t rowoff = ((size_t)b * Nn + n) * WCAT;
        kks[l]      = (float)P[rowoff + 3072 + qcol + l];
        kks[l + 32] = (float)P[rowoff + 3072 + qcol + l + 32];
        vks[l]      = (float)P[rowoff + 4096 + qcol + l];
        vks[l + 32] = (float)P[rowoff + 4096 + qcol + l + 32];
        qs[l]       = (float)P[rowoff + 0    + qcol + l];
        qs[l + 32]  = (float)P[rowoff + 0    + qcol + l + 32];
        __syncthreads();

        float kk0 = kks[l], kk1 = kks[l + 32];
        float sq = 0.f, y0 = 0.f, y1 = 0.f;
#pragma unroll
        for (int i = 0; i < 64; ++i) {
            float v = vks[i], q = qs[i];
            S0[i] += kk0 * v;
            S1[i] += kk1 * v;
            sq += S0[i] * S0[i] + S1[i] * S1[i];
            y0 += q * S0[i];
            y1 += q * S1[i];
        }
#pragma unroll
        for (int m = 16; m >= 1; m >>= 1) sq += __shfl_xor(sq, m, 32);
        float inv = 1.0f / (sqrtf(sq) + 1e-8f);

        size_t ob = ((size_t)bh * Nn + n) * 64;
        Qn[ob + l]      = (_Float16)(y0 * inv);
        Qn[ob + l + 32] = (_Float16)(y1 * inv);
        __syncthreads();
    }
}

// ---------------------------------------------------------------------------
// Kernel 4a/4b: rotary embedding + head-major relayout
// idx over B*H*N*32 pairs
// ---------------------------------------------------------------------------
__global__ void rotary_q_kernel(const _Float16* __restrict__ Qn,
                                _Float16* __restrict__ Qr) {
    int idx = blockIdx.x * blockDim.x + threadIdx.x;
    if (idx >= Bb * Hh * Nn * 32) return;
    int i2 = idx & 31;
    int n  = (idx >> 5) & (Nn - 1);
    int bh = idx >> 16;
    size_t base = ((size_t)bh * Nn + n) * 64 + 2 * i2;
    float x1 = (float)Qn[base], x2 = (float)Qn[base + 1];
    float freq = __powf(10000.f, -(float)(2 * i2) * (1.0f / 64.f));
    float a = (float)n * freq;
    float c = __cosf(a), s = __sinf(a);
    Qr[base]     = (_Float16)(x1 * c - x2 * s);
    Qr[base + 1] = (_Float16)(x2 * c + x1 * s);
}

__global__ void prep_kv_kernel(const _Float16* __restrict__ P,
                               _Float16* __restrict__ Kr,
                               _Float16* __restrict__ Vh) {
    int idx = blockIdx.x * blockDim.x + threadIdx.x;
    if (idx >= Bb * Hh * Nn * 32) return;
    int i2 = idx & 31;
    int n  = (idx >> 5) & (Nn - 1);
    int bh = idx >> 16;
    int h = bh & 15, b = bh >> 4;
    size_t prow = ((size_t)b * Nn + n) * WCAT;
    size_t base = ((size_t)bh * Nn + n) * 64 + 2 * i2;
    // K rotary
    float x1 = (float)P[prow + 1024 + h * 64 + 2 * i2];
    float x2 = (float)P[prow + 1024 + h * 64 + 2 * i2 + 1];
    float freq = __powf(10000.f, -(float)(2 * i2) * (1.0f / 64.f));
    float a = (float)n * freq;
    float c = __cosf(a), s = __sinf(a);
    Kr[base]     = (_Float16)(x1 * c - x2 * s);
    Kr[base + 1] = (_Float16)(x2 * c + x1 * s);
    // V relayout
    Vh[base]     = P[prow + 2048 + h * 64 + 2 * i2];
    Vh[base + 1] = P[prow + 2048 + h * 64 + 2 * i2 + 1];
}

// ---------------------------------------------------------------------------
// Kernel 5: causal flash attention. Grid (qtile=32, bh=32), 128 threads
// = 4 waves x 16 q-rows. K tile staged by TDM (async, padded rows); V tile
// staged manually transposed. QK^T and PV via WMMA; online softmax with
// width-16 shuffles matching the C-matrix lane layout. Output Oh f16,
// row-major [B*N, H*HD].
// ---------------------------------------------------------------------------
__global__ __launch_bounds__(128) void flash_attn_kernel(
    const _Float16* __restrict__ Qr, const _Float16* __restrict__ Kr,
    const _Float16* __restrict__ Vh, _Float16* __restrict__ Oh) {
    __shared__ _Float16 Ks[64][72];       // [kcol][d], 64 data + 8 pad halves
    __shared__ _Float16 Vst[64][72];      // [d][krow] (transposed)
    __shared__ _Float16 Ps[4][16][72];    // per-wave P tile [qrow][kcol]

    const int bh = blockIdx.y;
    const int qt = blockIdx.x;
    const int t = threadIdx.x, l = t & 31, w = t >> 5;
    const int lh = l >> 4, ll = l & 15;
    const size_t seqBase = (size_t)bh * Nn * 64;

    // Q fragments (A layout): row = ll, K(=d) chunks per half
    v16h qf[2];
    {
        const _Float16* qp = Qr + seqBase + (size_t)(qt * 64 + w * 16 + ll) * 64;
#pragma unroll
        for (int c = 0; c < 2; ++c) {
            const _Float16* p = qp + c * 32 + lh * 8;
            qf[c] = ldfrag_split(p, p + 16);
        }
    }

    v8f oacc[4];
#pragma unroll
    for (int dt = 0; dt < 4; ++dt)
#pragma unroll
        for (int r = 0; r < 8; ++r) oacc[dt][r] = 0.0f;
    float mrow[8], lrow[8];
#pragma unroll
    for (int r = 0; r < 8; ++r) { mrow[r] = -1e30f; lrow[r] = 0.0f; }

    const int skrow = t >> 1, sd0 = (t & 1) * 32;   // staging coords

    for (int kt = 0; kt <= qt; ++kt) {
        // --- stage K via TDM: 64 x 64-half tile, 8-half row padding ---
#if HAVE_TDM
        if (w == 0) {
            tdm_load_2d_f16((unsigned)(uintptr_t)&Ks[0][0],
                            Kr + seqBase + (size_t)(kt * 64) * 64,
                            /*tensor_d0=*/64, /*tensor_d1=*/Nn,
                            /*stride_d0=*/64,
                            /*tile_d0=*/64, /*tile_d1=*/64,
                            /*pad_interval: 32 dwords ->*/4,
                            /*pad_amount:   4 dwords ->*/3);
        }
#else
        {
            const _Float16* kp = Kr + seqBase + (size_t)(kt * 64 + skrow) * 64 + sd0;
#pragma unroll
            for (int c = 0; c < 4; ++c)
                *(v8h*)&Ks[skrow][sd0 + c * 8] = *(const v8h*)(kp + c * 8);
        }
#endif
        // --- stage V transposed (manual; TDM cannot transpose) ---
        {
            const _Float16* vp = Vh + seqBase + (size_t)(kt * 64 + skrow) * 64 + sd0;
#pragma unroll
            for (int c = 0; c < 4; ++c) {
                v8h vv = *(const v8h*)(vp + c * 8);
#pragma unroll
                for (int j = 0; j < 8; ++j) Vst[sd0 + c * 8 + j][skrow] = vv[j];
            }
        }
#if HAVE_TDM
        if (w == 0) __builtin_amdgcn_s_wait_tensorcnt(0);
#endif
        __syncthreads();

        // --- scores: 16 q-rows x 64 k-cols ---
        v8f sacc[4];
#pragma unroll
        for (int nt = 0; nt < 4; ++nt) {
            v8f s;
#pragma unroll
            for (int r = 0; r < 8; ++r) s[r] = 0.0f;
#pragma unroll
            for (int c = 0; c < 2; ++c) {
                const _Float16* p = &Ks[nt * 16 + ll][c * 32 + lh * 8];
                v16h kf = ldfrag_split(p, p + 16);
                s = wmma16(qf[c], kf, s);
            }
            sacc[nt] = s;
        }

        // --- online softmax ---
        float fscale[8];
#pragma unroll
        for (int r = 0; r < 8; ++r) {
            int qrow = qt * 64 + w * 16 + r + 8 * lh;
            float rmax = -1e30f;
#pragma unroll
            for (int nt = 0; nt < 4; ++nt) {
                int kcol = kt * 64 + nt * 16 + ll;
                float s = sacc[nt][r] * 0.125f;            // 1/sqrt(64)
                if (kcol > qrow) s = -1e30f;               // causal mask
                sacc[nt][r] = s;
                rmax = fmaxf(rmax, s);
            }
#pragma unroll
            for (int m = 1; m <= 8; m <<= 1)
                rmax = fmaxf(rmax, __shfl_xor(rmax, m, 16));
            float mnew = fmaxf(mrow[r], rmax);
            float corr = __expf(mrow[r] - mnew);
            float psum = 0.0f;
#pragma unroll
            for (int nt = 0; nt < 4; ++nt) {
                float p = __expf(sacc[nt][r] - mnew);
                sacc[nt][r] = p;
                psum += p;
            }
#pragma unroll
            for (int m = 1; m <= 8; m <<= 1) psum += __shfl_xor(psum, m, 16);
            lrow[r] = lrow[r] * corr + psum;
            mrow[r] = mnew;
            fscale[r] = corr;
        }
#pragma unroll
        for (int dt = 0; dt < 4; ++dt)
#pragma unroll
            for (int r = 0; r < 8; ++r) oacc[dt][r] *= fscale[r];

        // --- P: C layout -> LDS -> A layout ---
#pragma unroll
        for (int nt = 0; nt < 4; ++nt)
#pragma unroll
            for (int r = 0; r < 8; ++r)
                Ps[w][r + 8 * lh][nt * 16 + ll] = (_Float16)sacc[nt][r];
        __syncthreads();

        // --- O += P x V ---
#pragma unroll
        for (int kc = 0; kc < 2; ++kc) {
            const _Float16* pp = &Ps[w][ll][kc * 32 + lh * 8];
            v16h pf = ldfrag_split(pp, pp + 16);
#pragma unroll
            for (int dt = 0; dt < 4; ++dt) {
                const _Float16* vp = &Vst[dt * 16 + ll][kc * 32 + lh * 8];
                v16h vf = ldfrag_split(vp, vp + 16);
                oacc[dt] = wmma16(pf, vf, oacc[dt]);
            }
        }
        __syncthreads();
    }

    // --- epilogue: O /= l, write f16 row-major [B*N, H*64] ---
    const int b = bh >> 4, h = bh & 15;
#pragma unroll
    for (int dt = 0; dt < 4; ++dt)
#pragma unroll
        for (int r = 0; r < 8; ++r) {
            int qrow = qt * 64 + w * 16 + r + 8 * lh;
            int col = h * 64 + dt * 16 + ll;
            float v = oacc[dt][r] / lrow[r];
            Oh[(size_t)(b * Nn + qrow) * (Hh * HD_) + col] = (_Float16)v;
        }
}

// ---------------------------------------------------------------------------
// launcher
// ---------------------------------------------------------------------------
extern "C" void kernel_launch(void* const* d_in, const int* in_sizes, int n_in,
                              void* d_out, int out_size, void* d_ws, size_t ws_size,
                              hipStream_t stream) {
    const float* X     = (const float*)d_in[0];
    const float* wqkv  = (const float*)d_in[1];
    const float* wkvkv = (const float*)d_in[2];
    const float* wout  = (const float*)d_in[3];
    const float* bout  = (const float*)d_in[4];

    char* ws = (char*)d_ws;
    // workspace layout (bytes, 256-aligned; total ~100 MB)
    _Float16* Xh = (_Float16*)(ws + 0);               //  8,388,608
    _Float16* Wh = (_Float16*)(ws + 8388608);         // 10,485,760
    _Float16* Wo = (_Float16*)(ws + 18874368);        //  2,097,152
    _Float16* P  = (_Float16*)(ws + 20971520);        // 41,943,040 [4096 x 5120]
    _Float16* Qn = (_Float16*)(ws + 62914560);        //  8,388,608 [B,H,N,64]
    _Float16* Qr = (_Float16*)(ws + 71303168);        //  8,388,608
    _Float16* Kr = (_Float16*)(ws + 79691776);        //  8,388,608
    _Float16* Vh = (_Float16*)(ws + 88080384);        //  8,388,608
    _Float16* Oh = (_Float16*)(ws + 96468992);        //  8,388,608 [4096 x 1024]

    // 1. pack + convert to f16
    convert_pack_kernel<<<4096, 256, 0, stream>>>(X, wqkv, wkvkv, wout, Xh, Wh, Wo);

    // 2. fused projection GEMM: P = Xh[4096x1024] x Wh[1024x5120]
    gemm_wmma_kernel<true, false><<<dim3(WCAT / 128, BN / 128), 256, 0, stream>>>(
        Xh, Wh, (void*)P, nullptr, BN, WCAT, Dd, Dd, WCAT, WCAT);

    // 3. token-serial linear-attention scan -> Qn
    chunked_scan_kernel<<<Bb * Hh, 32, 0, stream>>>(P, Qn);

    // 4. rotary on Qn and K; relayout V to head-major
    rotary_q_kernel<<<(Bb * Hh * Nn * 32) / 256, 256, 0, stream>>>(Qn, Qr);
    prep_kv_kernel<<<(Bb * Hh * Nn * 32) / 256, 256, 0, stream>>>(P, Kr, Vh);

    // 5. causal flash attention -> Oh
    flash_attn_kernel<<<dim3(Nn / 64, Bb * Hh), 128, 0, stream>>>(Qr, Kr, Vh, Oh);

    // 6. output projection + bias -> d_out (f32)
    gemm_wmma_kernel<false, true><<<dim3(Dd / 128, BN / 128), 256, 0, stream>>>(
        Oh, Wo, d_out, bout, BN, Dd, Dd, Dd, Dd, Dd);
}
